// LorentzPureConv_21199958573229
// MI455X (gfx1250) — matrix operand order repaired and checked by reading, not compiled, not run on hardware
//
#include <hip/hip_runtime.h>
#include <hip/hip_bf16.h>

// ---- problem constants -----------------------------------------------------
constexpr int Bn    = 32;
constexpr int Himg  = 64;
constexpr int Wimg  = 64;
constexpr int C_IN  = 65;          // channel 0 = time-like, 1..64 space-like
constexpr int C_OUT = 129;
constexpr int M_TOT = Bn * Himg * Wimg;      // 131072 rows
constexpr int K_RAW = (C_IN - 1) * 9 + 1;    // 577
constexpr int KP    = 608;                   // padded K, 19 steps of 32
constexpr int KSTEP = KP / 32;               // 19
constexpr int N_EFF = 128;                   // weight rows 1..128 (row 0 of v is discarded)

typedef __attribute__((ext_vector_type(16))) _Float16 v16h;
typedef __attribute__((ext_vector_type(8)))  _Float16 v8h;
typedef __attribute__((ext_vector_type(8)))  float    v8f;

union HFrag { v16h v; v8h h[2]; };

// ---- kernel 0: weight[1:129, 0:577] -> f16 (128 x 608, zero-padded) --------
__global__ void convert_weight(const float* __restrict__ w, _Float16* __restrict__ wf) {
    int idx = blockIdx.x * blockDim.x + threadIdx.x;
    if (idx >= N_EFF * KP) return;
    int n = idx / KP;
    int k = idx - n * KP;
    float v = (k < K_RAW) ? w[(size_t)(n + 1) * K_RAW + k] : 0.0f;
    wf[idx] = (_Float16)v;
}

// ---- kernel 1: Lorentz im2col + normalization -> f16 rows (M x 608) --------
// one wave32 per output pixel; lane owns space channels (lane) and (lane+32)
__global__ __launch_bounds__(256) void preprocess(const float* __restrict__ x,
                                                  _Float16* __restrict__ A) {
    const int lane = threadIdx.x & 31;
    const int m    = blockIdx.x * 8 + (threadIdx.x >> 5);   // pixel/row index
    if (m >= M_TOT) return;

    const int b   = m >> 12;           // / (64*64)
    const int pix = m & 4095;
    const int oy  = pix >> 6;
    const int ox  = pix & 63;

    const float* xb  = x + (size_t)b * Himg * Wimg * C_IN;
    _Float16*    row = A + (size_t)m * KP;

    float tsq_sum = 0.0f;
    #pragma unroll
    for (int k = 0; k < 9; ++k) {
        const int iy  = oy + (k / 3) - 1;
        const int ix  = ox + (k % 3) - 1;
        const bool in = (iy >= 0) & (iy < Himg) & (ix >= 0) & (ix < Wimg);
        const size_t off = in ? ((size_t)iy * Wimg + ix) * C_IN : 0;

        float p0  = in ? xb[off]             : 0.0f;   // uniform broadcast load
        float s_a = in ? xb[off + 1 + lane]  : 0.0f;
        float s_b = in ? xb[off + 33 + lane] : 0.0f;

        float t  = fmaxf(p0, 1.0f);                    // clip(p0, min=1)
        float ss = s_a * s_a + s_b * s_b;
        #pragma unroll
        for (int d = 16; d >= 1; d >>= 1) ss += __shfl_xor(ss, d, 32);

        float max_norm = sqrtf(fmaxf(t * t - 1.0f, 1e-8f));
        float s_norm   = sqrtf(ss + 1e-8f);
        float scale    = max_norm / s_norm;

        row[1 + k * 64 + lane]      = (_Float16)(s_a * scale);
        row[1 + k * 64 + lane + 32] = (_Float16)(s_b * scale);
        tsq_sum += t * t;                              // uniform across lanes
    }
    if (lane == 0) row[0] = (_Float16)sqrtf(tsq_sum - 8.0f);  // t_merged
    if (lane >= 1) row[576 + lane] = (_Float16)0.0f;          // pad 577..607
}

// ---- kernel 2: f16 WMMA GEMM (M x 608) x (608 x 128) + Lorentz epilogue ----
// 8 waves/block; wave w owns rows [blk*128 + 16w, +16) and ALL 8 N-tiles,
// so the sqrt(sum v^2 + 1) epilogue is a 16-lane shfl reduction.
__global__ __launch_bounds__(256) void lorentz_gemm(const _Float16* __restrict__ A,
                                                    const _Float16* __restrict__ Wm,
                                                    const float*    __restrict__ bias,
                                                    float*          __restrict__ out) {
    __shared__ __align__(16) _Float16 sA[128 * 32];
    __shared__ __align__(16) _Float16 sB[128 * 32];

    const int tid    = threadIdx.x;
    const int wave   = tid >> 5;
    const int lane   = tid & 31;
    const int mblock = blockIdx.x * 128;
    const int m0     = mblock + wave * 16;

    v8f acc[8];
    #pragma unroll
    for (int i = 0; i < 8; ++i)
        acc[i] = (v8f){0.f, 0.f, 0.f, 0.f, 0.f, 0.f, 0.f, 0.f};

    for (int ks = 0; ks < KSTEP; ++ks) {
        __syncthreads();                       // previous tile fully consumed
        const int kbase = ks * 32;
        // stage 128x32 A-slice and 128x32 B-slice (8 KB each, 16 B per copy)
        #pragma unroll
        for (int r = 0; r < 2; ++r) {
            const int c   = tid + r * 256;     // 512 chunks of 8 halves
            const int row = c >> 2;
            const int off = (c & 3) * 8;
            *(v8h*)&sA[row * 32 + off] =
                *(const v8h*)&A[(size_t)(mblock + row) * KP + kbase + off];
            *(v8h*)&sB[row * 32 + off] =
                *(const v8h*)&Wm[(size_t)row * KP + kbase + off];
        }
        __syncthreads();

        // A fragment: lanes 0-15 -> K 0..7 & 16..23 ; lanes 16-31 -> 8..15 & 24..31
        HFrag a;
        {
            const int row = wave * 16 + (lane & 15);
            const int off = (lane >> 4) << 3;
            a.h[0] = *(const v8h*)&sA[row * 32 + off];
            a.h[1] = *(const v8h*)&sA[row * 32 + off + 16];
        }
        // B fragments: lane holds N = lane&15; lanes 0-15 -> K 0..7 & 16..23 ; lanes 16-31 -> 8..15 & 24..31
        #pragma unroll
        for (int nt = 0; nt < 8; ++nt) {
            HFrag bf;
            const int row  = nt * 16 + (lane & 15);   // N index (Wm is N-major)
            const int boff = (lane >> 4) << 3;
            bf.h[0] = *(const v8h*)&sB[row * 32 + boff];
            bf.h[1] = *(const v8h*)&sB[row * 32 + boff + 16];
            acc[nt] = __builtin_amdgcn_wmma_f32_16x16x32_f16(
                false, a.v, false, bf.v, (short)0, acc[nt], false, false);
        }
    }

    // epilogue: v = acc + bias[1+col]; out[:,0] = sqrt(sum v^2 + 1); out[:,1+col] = v
    const int ncol = lane & 15;
    const int hsel = lane >> 4;                 // 0: M=e, 1: M=8+e
    float bv[8];
    #pragma unroll
    for (int nt = 0; nt < 8; ++nt) bv[nt] = bias[1 + nt * 16 + ncol];

    #pragma unroll
    for (int e = 0; e < 8; ++e) {
        const int row = m0 + hsel * 8 + e;
        float vals[8];
        float sumsq = 0.0f;
        #pragma unroll
        for (int nt = 0; nt < 8; ++nt) {
            float v = acc[nt][e] + bv[nt];
            vals[nt] = v;
            sumsq += v * v;
        }
        sumsq += __shfl_xor(sumsq, 1, 32);      // reduce within 16-lane half
        sumsq += __shfl_xor(sumsq, 2, 32);
        sumsq += __shfl_xor(sumsq, 4, 32);
        sumsq += __shfl_xor(sumsq, 8, 32);
        const float vt = sqrtf(sumsq + 1.0f);

        float* orow = out + (size_t)row * C_OUT;
        if (ncol == 0) orow[0] = vt;
        #pragma unroll
        for (int nt = 0; nt < 8; ++nt) orow[1 + nt * 16 + ncol] = vals[nt];
    }
}

// ---- host launcher ---------------------------------------------------------
extern "C" void kernel_launch(void* const* d_in, const int* in_sizes, int n_in,
                              void* d_out, int out_size, void* d_ws, size_t ws_size,
                              hipStream_t stream) {
    const float* x    = (const float*)d_in[0];
    const float* w    = (const float*)d_in[1];
    const float* bias = (const float*)d_in[2];
    float*       out  = (float*)d_out;

    // workspace: A_f16 (M x KP) then W_f16 (128 x KP); both 16B-aligned
    _Float16* Af = (_Float16*)d_ws;
    _Float16* Wf = (_Float16*)((char*)d_ws + (size_t)M_TOT * KP * sizeof(_Float16));

    convert_weight<<<(N_EFF * KP + 255) / 256, 256, 0, stream>>>(w, Wf);
    preprocess<<<M_TOT / 8, 256, 0, stream>>>(x, Af);
    lorentz_gemm<<<M_TOT / 128, 256, 0, stream>>>(Af, Wf, bias, out);
}
